// OrthogonalChannelMixing_23218593202195
// MI455X (gfx1250) — compile-verified, matrix operand order checked by reading, not run on hardware
//
#include <hip/hip_runtime.h>

// CDNA5 / gfx1250, wave32.
// Pipeline:
//   1) prep:    Aug[512x1024] = [I+A | I-A],  A = W - W^T
//   2) 16x { gj_panel (1 WG, 32 pivots, product-form transform Tc + pivot-row copy Prow)
//            gj_update (grid rank-32 update of Aug, LDS-tiled, async-to-LDS fills) }
//      -> Aug[:,512:] = K = (I+A)^-1 (I-A)
//   3) pack:    K -> f16 hi/lo in WMMA A-operand lane layout
//   4) gemm:    y[b] = K * x[b] via v_wmma_f32_16x16x32_f16, split-f16 (3 products),
//               double-buffered LDS, x converted fp32->f16 hi/lo in registers and
//               scattered with b64 DS stores into the B-operand layout.
// Workspace: 512*1024 + 512*32 + 32*1024 floats + 2*512*512 halves = ~3.2 MB.

typedef __attribute__((ext_vector_type(16))) _Float16 v16h;
typedef __attribute__((ext_vector_type(4)))  _Float16 v4h;
typedef __attribute__((ext_vector_type(8)))  float    v8f;
typedef __attribute__((ext_vector_type(4)))  int      v4i;

#define CCH 512

#if defined(__has_builtin)
#if __has_builtin(__builtin_amdgcn_global_load_async_to_lds_b128) && \
    __has_builtin(__builtin_amdgcn_s_wait_asynccnt)
#define USE_ASYNC_LDS 1
// param types per hipcc diagnostic: (int4 __device__*, int4 __shared__*, imm, imm)
typedef __attribute__((address_space(1))) v4i* gas_p;
typedef __attribute__((address_space(3))) v4i* las_p;
#endif
#endif

// ---------------- 1) build augmented system [I+A | I-A] ----------------
__global__ __launch_bounds__(256) void prep_kernel(const float* __restrict__ W,
                                                   float* __restrict__ Aug) {
  int g  = blockIdx.x * 256 + threadIdx.x;       // 0..262143
  int r  = g >> 9;
  int c  = g & 511;
  float a = W[r * CCH + c] - W[c * CCH + r];
  float d = (r == c) ? 1.0f : 0.0f;
  Aug[(size_t)r * 1024 + c]       = d + a;
  Aug[(size_t)r * 1024 + 512 + c] = d - a;
}

// ---------------- 2a) panel: 32 Gauss-Jordan pivots, record transform ----------------
// Thread r owns row r of the 512x32 panel; fully unrolled so P/T stay in VGPRs.
// No pivoting needed: sym(I+A) = I > 0 guarantees stability.
__global__ __launch_bounds__(512) void gj_panel(float* __restrict__ Aug,
                                                float* __restrict__ Tc,
                                                float* __restrict__ Prow,
                                                int kb) {
  const int r = threadIdx.x;  // 0..511
  // snapshot the 32 pivot rows (pre-round values) for the update kernel
  for (int idx = r; idx < 32 * 1024; idx += 512)
    Prow[idx] = Aug[(size_t)(kb + (idx >> 10)) * 1024 + (idx & 1023)];
  __syncthreads();

  float P[32], T[32];
#pragma unroll
  for (int q = 0; q < 32; ++q) {
    P[q] = Aug[(size_t)r * 1024 + kb + q];
    T[q] = (r == kb + q) ? 1.0f : 0.0f;   // I-slab columns kb..kb+31
  }

  __shared__ float pivP[32], pivT[32];
#pragma unroll
  for (int p = 0; p < 32; ++p) {
    if (r == kb + p) {
      float inv = 1.0f / P[p];
#pragma unroll
      for (int q = 0; q < 32; ++q) {
        P[q] *= inv; T[q] *= inv;
        pivP[q] = P[q]; pivT[q] = T[q];
      }
    }
    __syncthreads();
    if (r != kb + p) {
      float f = P[p];
#pragma unroll
      for (int q = 0; q < 32; ++q) {       // pivP[q<p]==0, so full loop is exact
        P[q] = fmaf(-f, pivP[q], P[q]);
        T[q] = fmaf(-f, pivT[q], T[q]);
      }
    }
    __syncthreads();
  }
#pragma unroll
  for (int q = 0; q < 32; ++q) Tc[r * 32 + q] = T[q];
}

// ---------------- 2b) rank-32 product-form update: A <- T*A on cols >= kb ----------------
__global__ __launch_bounds__(256) void gj_update(float* __restrict__ Aug,
                                                 const float* __restrict__ Tc,
                                                 const float* __restrict__ Prow,
                                                 int kb) {
  const int c0 = blockIdx.x * 64;
  const int r0 = blockIdx.y * 64;
  if (c0 + 64 <= kb) return;               // tile entirely already-reduced
  __shared__ float Ts[64 * 32];
  __shared__ float Ps[32 * 64];
  const int tid = threadIdx.x;
#ifdef USE_ASYNC_LDS
  // gfx1250 async copy: global -> LDS without a VGPR round-trip (ASYNCcnt-tracked)
#pragma unroll
  for (int i = 0; i < 2; ++i) {
    int f4 = tid + i * 256;                // 512 x 16B = 64x32 floats
    __builtin_amdgcn_global_load_async_to_lds_b128(
        (gas_p)(Tc + (size_t)r0 * 32 + f4 * 4),
        (las_p)(&Ts[f4 * 4]), 0, 0);
  }
#pragma unroll
  for (int i = 0; i < 2; ++i) {
    int f4 = tid + i * 256;                // 32 rows x 16 float4
    __builtin_amdgcn_global_load_async_to_lds_b128(
        (gas_p)(Prow + (size_t)(f4 >> 4) * 1024 + c0 + (f4 & 15) * 4),
        (las_p)(&Ps[f4 * 4]), 0, 0);
  }
  __builtin_amdgcn_s_wait_asynccnt(0);
  __syncthreads();
#else
#pragma unroll
  for (int i = 0; i < 8; ++i) {
    int idx = tid + i * 256;
    Ts[idx] = Tc[(r0 + (idx >> 5)) * 32 + (idx & 31)];
  }
#pragma unroll
  for (int i = 0; i < 8; ++i) {
    int idx = tid + i * 256;
    Ps[idx] = Prow[(idx >> 6) * 1024 + c0 + (idx & 63)];
  }
  __syncthreads();
#endif

  const int tr = (tid >> 4) * 4;           // 4x4 micro-tile per thread
  const int tc = (tid & 15) * 4;
  float acc[4][4];
#pragma unroll
  for (int i = 0; i < 4; ++i) {
    int r = r0 + tr + i;
    bool piv = (r >= kb) && (r < kb + 32); // pivot rows are fully replaced by T*Prow
#pragma unroll
    for (int j = 0; j < 4; ++j) {
      int c = c0 + tc + j;
      acc[i][j] = (piv || c < kb) ? 0.0f : Aug[(size_t)r * 1024 + c];
    }
  }
#pragma unroll 8
  for (int p = 0; p < 32; ++p) {
    float a[4], bv[4];
#pragma unroll
    for (int i = 0; i < 4; ++i) a[i] = Ts[(tr + i) * 32 + p];
#pragma unroll
    for (int j = 0; j < 4; ++j) bv[j] = Ps[p * 64 + tc + j];
#pragma unroll
    for (int i = 0; i < 4; ++i)
#pragma unroll
      for (int j = 0; j < 4; ++j) acc[i][j] = fmaf(a[i], bv[j], acc[i][j]);
  }
#pragma unroll
  for (int i = 0; i < 4; ++i) {
    int r = r0 + tr + i;
#pragma unroll
    for (int j = 0; j < 4; ++j) {
      int c = c0 + tc + j;
      if (c >= kb) Aug[(size_t)r * 1024 + c] = acc[i][j];
    }
  }
}

// ---------------- 3) pack K into WMMA A-operand layout, split f16 hi/lo ----------------
// 16-bit A 16x32 layout (ISA 7.12.2): lane L<16 -> row=L, k = h + (h>=8?8:0)
//                                     lane L>=16 -> row=L-16, k = 8 + h + (h>=8?8:0)
__global__ __launch_bounds__(256) void pack_kernel(const float* __restrict__ Aug,
                                                   _Float16* __restrict__ Ahi,
                                                   _Float16* __restrict__ Alo) {
  int g = blockIdx.x * 256 + threadIdx.x;  // 0..262143
  int h    = g & 15;
  int lane = (g >> 4) & 31;
  int t    = g >> 9;
  int kt   = t & 15;
  int jt   = t >> 4;
  int j    = jt * 16 + (lane & 15);
  int base = (lane < 16) ? 0 : 8;
  int k    = kt * 32 + base + h + ((h >= 8) ? 8 : 0);
  float f = Aug[(size_t)j * 1024 + 512 + k];     // K = right half of reduced Aug
  _Float16 hi = (_Float16)f;
  _Float16 lo = (_Float16)(f - (float)hi);
  Ahi[g] = hi;
  Alo[g] = lo;
}

// ---------------- 4) main GEMM: y[b] = K * x[b] via split-f16 WMMA ----------------
// Block tile: 128(J) x 128(N), 256 threads = 8 waves.
// Wave w: A-row strip 16 rows AND x-load strip of 4 k-rows (k0 = w*4).
// Per thread: 4k x 4n micro-tile of x -> 4 coalesced b128 loads, 8 b64 DS stores
// (4 consecutive k land on 4 consecutive halves of the B operand).
// LDS double-buffered: one barrier per k-step; loads/convert of tile i+1 overlap WMMA of tile i.
__global__ __launch_bounds__(256) void gemm_kernel(const float* __restrict__ x,
                                                   const _Float16* __restrict__ Ahi,
                                                   const _Float16* __restrict__ Alo,
                                                   float* __restrict__ y) {
  const int n0   = blockIdx.x * 128;
  const int j0   = blockIdx.y * 128;
  const int b    = blockIdx.z;
  const int tid  = threadIdx.x;
  const int lane = tid & 31;
  const int wave = tid >> 5;

  __shared__ _Float16 BhiBuf[2][8 * 32 * 16];    // 2 x 8KB
  __shared__ _Float16 BloBuf[2][8 * 32 * 16];    // 2 x 8KB

  v8f zf = {};
  v8f acc[8];
#pragma unroll
  for (int s = 0; s < 8; ++s) acc[s] = zf;

  const int jt = (j0 >> 4) + wave;                       // 16-row A tile index
  const int k0 = wave * 4;                               // x k-row strip of this wave
  const int nl = (lane & 31) * 4;                        // 4-col group of this lane
  const float* xb = x + ((size_t)b * CCH) * 16384 + n0;  // x[b, :, n0..]

  // precomputed B-operand scatter geometry (k0 is a multiple of 4 -> 4 consecutive halves)
  const int Lgrp = (k0 & 8) ? 16 : 0;
  const int h0   = (k0 & 7) + ((k0 & 16) ? 8 : 0);

  float vv[4][4];
  auto loadTile = [&](int kk) {
#pragma unroll
    for (int i = 0; i < 4; ++i) {
      float4 f = *(const float4*)(xb + (size_t)(kk + k0 + i) * 16384 + nl);
      vv[i][0] = f.x; vv[i][1] = f.y; vv[i][2] = f.z; vv[i][3] = f.w;
    }
  };
  auto scatterTile = [&](int buf) {
#pragma unroll
    for (int j = 0; j < 4; ++j) {
      int n  = nl + j;
      int s  = n >> 4;
      int cc = n & 15;
      int idx = (((s * 32) + cc + Lgrp) << 4) + h0;   // half index, 8B aligned
      float e0 = vv[0][j], e1 = vv[1][j], e2 = vv[2][j], e3 = vv[3][j];
      v4h hi = {(_Float16)e0, (_Float16)e1, (_Float16)e2, (_Float16)e3};
      v4h lo = {(_Float16)(e0 - (float)hi.x), (_Float16)(e1 - (float)hi.y),
                (_Float16)(e2 - (float)hi.z), (_Float16)(e3 - (float)hi.w)};
      *(v4h*)(&BhiBuf[buf][idx]) = hi;
      *(v4h*)(&BloBuf[buf][idx]) = lo;
    }
  };

  loadTile(0);
  scatterTile(0);
  int cur = 0;

  for (int kk = 0; kk < CCH; kk += 32) {
    __syncthreads();                       // writes to buf[cur] visible to all waves
    const int kt = kk >> 5;
    v16h ahi = *(const v16h*)(Ahi + (((size_t)(jt * 16 + kt) * 32 + lane) << 4));
    v16h alo = *(const v16h*)(Alo + (((size_t)(jt * 16 + kt) * 32 + lane) << 4));
    const bool more = (kk + 32 < CCH);
    if (more) loadTile(kk + 32);           // overlap with WMMA below
    if (kk + 64 < CCH)                     // speculative prefetch 2 tiles ahead
      __builtin_prefetch(xb + (size_t)(kk + 64 + k0) * 16384 + nl, 0, 1);
#pragma unroll
    for (int s = 0; s < 8; ++s) {
      v16h bhi = *(const v16h*)&BhiBuf[cur][(s * 32 + lane) << 4];
      v16h blo = *(const v16h*)&BloBuf[cur][(s * 32 + lane) << 4];
      acc[s] = __builtin_amdgcn_wmma_f32_16x16x32_f16(false, ahi, false, bhi,
                                                      (short)0, acc[s], false, false);
      acc[s] = __builtin_amdgcn_wmma_f32_16x16x32_f16(false, ahi, false, blo,
                                                      (short)0, acc[s], false, false);
      acc[s] = __builtin_amdgcn_wmma_f32_16x16x32_f16(false, alo, false, bhi,
                                                      (short)0, acc[s], false, false);
    }
    if (more) scatterTile(cur ^ 1);        // safe: prior reads of buf[cur^1] fenced above
    cur ^= 1;
  }

  // ---- epilogue: C/D layout -> lane L: col = L&15, VGPR r: row = r + (L<16?0:8) ----
  const int jrow0 = j0 + wave * 16 + ((lane < 16) ? 0 : 8);
  const int ncol  = n0 + (lane & 15);
#pragma unroll
  for (int s = 0; s < 8; ++s)
#pragma unroll
    for (int r = 0; r < 8; ++r)
      y[((size_t)b * CCH + jrow0 + r) * 16384 + ncol + s * 16] = acc[s][r];
}

// ---------------- launch ----------------
extern "C" void kernel_launch(void* const* d_in, const int* in_sizes, int n_in,
                              void* d_out, int out_size, void* d_ws, size_t ws_size,
                              hipStream_t stream) {
  const float* x = (const float*)d_in[0];     // (16, 512, 16384) fp32
  const float* W = (const float*)d_in[1];     // (512, 512) fp32
  float* y = (float*)d_out;                   // (16, 512, 16384) fp32

  float*    Aug  = (float*)d_ws;              // 512*1024
  float*    Tcb  = Aug + 512 * 1024;          // 512*32
  float*    Prow = Tcb + 512 * 32;            // 32*1024
  _Float16* Ahi  = (_Float16*)(Prow + 32 * 1024);  // 512*512
  _Float16* Alo  = Ahi + 512 * 512;                // 512*512

  prep_kernel<<<1024, 256, 0, stream>>>(W, Aug);
  for (int kb = 0; kb < CCH; kb += 32) {
    gj_panel<<<1, 512, 0, stream>>>(Aug, Tcb, Prow, kb);
    gj_update<<<dim3(16, 8), 256, 0, stream>>>(Aug, Tcb, Prow, kb);
  }
  pack_kernel<<<1024, 256, 0, stream>>>(Aug, Ahi, Alo);
  gemm_kernel<<<dim3(128, 4, 16), 256, 0, stream>>>(x, Ahi, Alo, y);
}
